// BlockRecurrentSwinIRBlock_31293131719199
// MI455X (gfx1250) — matrix operand order repaired and checked by reading, not compiled, not run on hardware
//
#include <hip/hip_runtime.h>
#include <hip/hip_bf16.h>

// MI455X / gfx1250, wave32. All GEMMs via v_wmma_f32_16x16x32_f16.
// Weights are repacked once (pack_w_kernel) into exact WMMA-B fragment order:
//   dst[((kt*JT + jt)*32 + lane)*16 + e] = W[(kt*32 + 16*(lane>>4) + e)*N + jt*16 + (lane&15)]
// so each lane's B fragment is one contiguous 32-byte load.

typedef __attribute__((ext_vector_type(16))) _Float16 v16h;
typedef __attribute__((ext_vector_type(8)))  _Float16 v8h;
typedef __attribute__((ext_vector_type(8)))  float    v8f;

#define WMMA_F16(A_, B_, C_) \
  __builtin_amdgcn_wmma_f32_16x16x32_f16(false, (A_), false, (B_), (short)0, (C_), false, false)

static __device__ __forceinline__ v16h cat16(v8h lo, v8h hi) {
  v16h r;
#pragma unroll
  for (int i = 0; i < 8; ++i) { r[i] = lo[i]; r[i + 8] = hi[i]; }
  return r;
}
static __device__ __forceinline__ v8h zero8() {
  v8h z;
#pragma unroll
  for (int i = 0; i < 8; ++i) z[i] = (_Float16)0.0f;
  return z;
}

// ---- weight pack: f32 row-major (192 x ncols) -> f16 WMMA-B fragment order ----
__global__ __launch_bounds__(256) void pack_w_kernel(const float* __restrict__ W,
                                                     _Float16* __restrict__ dst, int ncols) {
  int jtc   = ncols >> 4;            // N tiles
  int total = 6 * jtc * 512;         // 6 K-tiles x jtc x 32 lanes x 16 elems
  int t = blockIdx.x * 256 + threadIdx.x;
  if (t >= total) return;
  int e    = t & 15;
  int lane = (t >> 4) & 31;
  int tile = t >> 9;
  int jt = tile % jtc;
  int kt = tile / jtc;
  int row = kt * 32 + ((lane >> 4) << 4) + e;   // K index (contiguous k, same as A: lanes16-31 hold K=16..31)
  int col = (jt << 4) + (lane & 15);            // N index
  dst[t] = (_Float16)W[row * ncols + col];
}

// ---- fused per-window kernel: embed GEMMs -> 16 attention heads -> projections ----
__global__ __launch_bounds__(256) void swin_attn_kernel(
    const float* __restrict__ input_x,  const float* __restrict__ state_x,
    const float* __restrict__ bias_xs,
    const float* __restrict__ bt_cross_s, const float* __restrict__ bt_self_s,
    const float* __restrict__ bt_cross_x, const float* __restrict__ bt_self_x,
    const float* __restrict__ b_state,  const float* __restrict__ b_embed,
    const float* __restrict__ b_projx,  const float* __restrict__ b_projs,
    const _Float16* __restrict__ wpack, float* __restrict__ out) {
  __shared__ _Float16 smem[81920];   // 160 KB -> 2 workgroups / WGP
  _Float16* sh = smem;               // 64x192 state f16 ; reused as ss (state attn out)
  _Float16* xh = smem + 12288;       // 64x192 x f16     ; reused as ax (x attn out)
  _Float16* sl = smem + 24576;       // 64x384 state embeddings (k|v|q|qsx)
  _Float16* el = smem + 49152;       // 64x384 x embeddings     (k|v|q|qxs)
  _Float16* pb = smem + 73728;       // 8 waves x (16x64) softmax probs scratch

  const int tid  = threadIdx.x;
  const int lane = tid & 31;
  const int wv   = tid >> 5;
  const int hl   = lane >> 4;        // wave half (C-layout: M = r + 8*hl)
  const int l15  = lane & 15;
  const size_t ibase = (size_t)blockIdx.x * (64 * 192);

  // ---- Phase 1: stage inputs to LDS (f16); x gets per-token additive bias ----
  {
    const float4* s4 = (const float4*)(state_x + ibase);
    const float4* x4 = (const float4*)(input_x + ibase);
    for (int i = tid; i < 3072; i += 256) {
      float4 sv = s4[i];
      float4 xv = x4[i];
      float tb = bias_xs[i / 48];    // token = (i*4)/192
      int o = i << 2;
      sh[o]   = (_Float16)sv.x;        sh[o+1] = (_Float16)sv.y;
      sh[o+2] = (_Float16)sv.z;        sh[o+3] = (_Float16)sv.w;
      xh[o]   = (_Float16)(xv.x + tb); xh[o+1] = (_Float16)(xv.y + tb);
      xh[o+2] = (_Float16)(xv.z + tb); xh[o+3] = (_Float16)(xv.w + tb);
    }
  }
  __syncthreads();

  // ---- Phase 2: sl = sh@W_state+b ; el = xh@W_embed+b  (64x192 @ 192x384) ----
  for (int t = wv; t < 192; t += 8) {
    int g  = (t >= 96);
    int tl = t - 96 * g;
    int mt = tl / 24, jt = tl % 24;
    const _Float16* A  = g ? xh : sh;
    const _Float16* Bp = wpack + (g ? 73728 : 0);
    const float*    bi = g ? b_embed : b_state;
    _Float16*       D  = g ? el : sl;
    v8f acc = {};
#pragma unroll
    for (int kt = 0; kt < 6; ++kt) {
      const _Float16* ap = A + (mt * 16 + l15) * 192 + kt * 32 + 16 * hl;
      v16h af = cat16(*(const v8h*)ap, *(const v8h*)(ap + 8));      // A: K = 16h + e (contiguous)
      const _Float16* bp = Bp + (((kt * 24 + jt) * 32 + lane) << 4);
      v16h bf = cat16(*(const v8h*)bp, *(const v8h*)(bp + 8));      // packed B fragment
      acc = WMMA_F16(af, bf, acc);
    }
    float bj = bi[(jt << 4) + l15];
#pragma unroll
    for (int r = 0; r < 8; ++r)
      D[(mt * 16 + r + 8 * hl) * 384 + (jt << 4) + l15] = (_Float16)(acc[r] + bj);
  }
  __syncthreads();

  // ---- Phase 3: 16 (branch, head) attention problems, 2 per wave ----
  const float scale = 0.2041241452319315f;   // 24^-0.5
  for (int id = wv; id < 16; id += 8) {
    int br = id >> 2;                         // 0:self_s 1:cross_s 2:self_x 3:cross_x
    int h  = id & 3;
    const _Float16* qb = (br < 2) ? sl : el;  // qs,qsx from state; qx,qxs from x
    int qOff = (br & 1) ? 288 : 192;
    const bool sKV = (br == 0) || (br == 3);
    const _Float16* kb = sKV ? sl : el;
    const _Float16* vb = sKV ? sl : el;
    const float* tab = (br == 0) ? bt_self_s : (br == 1) ? bt_cross_s
                     : (br == 2) ? bt_self_x : bt_cross_x;
    _Float16* ob = (br < 2) ? sh : xh;        // ss / ax (region reuse)
    int oOff = ((br & 1) ? 96 : 0) + h * 24;
    _Float16* P = pb + wv * 1024;

    // Preload K^T B-fragments: lane = key token, elems = head-dim (pad 24->32 with zeros)
    v16h kf[4];
#pragma unroll
    for (int nt = 0; nt < 4; ++nt) {
      const _Float16* kp = kb + (nt * 16 + l15) * 384 + h * 24 + 16 * hl;
      v8h k0 = *(const v8h*)kp;
      v8h k1 = hl ? zero8() : *(const v8h*)(kp + 8);   // d>=24 -> 0
      kf[nt] = cat16(k0, k1);
    }

    // Preload the four V B-fragments ([dt][kt]) -- invariant across row blocks.
    // lane -> output head-dim column d, elems -> key token (strided LDS gather).
    v16h vfrag[2][2];
#pragma unroll
    for (int dt = 0; dt < 2; ++dt) {
      int d = dt * 16 + l15;
      const _Float16* vcol = vb + 96 + h * 24 + d;
#pragma unroll
      for (int kt = 0; kt < 2; ++kt) {
        v16h vf;
#pragma unroll
        for (int e = 0; e < 16; ++e)
          vf[e] = vcol[(kt * 32 + 16 * hl + e) * 384];
        vfrag[dt][kt] = vf;
      }
    }

    // Rolled row-block loop: keeps live VGPRs <= 256 (no VGPR-MSB mode churn),
    // preserving 4 waves/SIMD alongside the 160 KB LDS footprint.
    for (int mt = 0; mt < 4; ++mt) {
      // Q A-fragment (K = head-dim, zero-padded)
      const _Float16* qp = qb + (mt * 16 + l15) * 384 + qOff + h * 24 + 16 * hl;
      v8h q0 = *(const v8h*)qp;
      v8h q1 = hl ? zero8() : *(const v8h*)(qp + 8);   // d>=24 -> 0
      v16h qf = cat16(q0, q1);

      v8f sA[4];
#pragma unroll
      for (int nt = 0; nt < 4; ++nt) { v8f z = {}; sA[nt] = WMMA_F16(qf, kf[nt], z); }

      // scale + relative-position bias (closed-form index, tables L2-resident)
#pragma unroll
      for (int nt = 0; nt < 4; ++nt)
#pragma unroll
        for (int r = 0; r < 8; ++r) {
          int nq = mt * 16 + r + 8 * hl;
          int mk = nt * 16 + l15;
          int idx = (((nq >> 3) - (mk >> 3)) + 7) * 15 + ((nq & 7) - (mk & 7)) + 7;
          sA[nt][r] = scale * sA[nt][r] + tab[idx * 4 + h];
        }

      // row softmax: reduce over 4 tiles + 16 lanes of this half-wave
#pragma unroll
      for (int r = 0; r < 8; ++r) {
        float m = fmaxf(fmaxf(sA[0][r], sA[1][r]), fmaxf(sA[2][r], sA[3][r]));
        m = fmaxf(m, __shfl_xor(m, 1, 32));
        m = fmaxf(m, __shfl_xor(m, 2, 32));
        m = fmaxf(m, __shfl_xor(m, 4, 32));
        m = fmaxf(m, __shfl_xor(m, 8, 32));
        float sum = 0.f;
#pragma unroll
        for (int nt = 0; nt < 4; ++nt) {
          float e = __expf(sA[nt][r] - m);
          sA[nt][r] = e; sum += e;
        }
        sum += __shfl_xor(sum, 1, 32);
        sum += __shfl_xor(sum, 2, 32);
        sum += __shfl_xor(sum, 4, 32);
        sum += __shfl_xor(sum, 8, 32);
        float inv = 1.0f / sum;
        int prow = (r + 8 * hl) * 64 + l15;
#pragma unroll
        for (int nt = 0; nt < 4; ++nt) P[prow + nt * 16] = (_Float16)(sA[nt][r] * inv);
      }

      // P @ V for this 16-row block (K = 64 tokens = 2 WMMA k-steps)
#pragma unroll
      for (int dt = 0; dt < 2; ++dt) {
        v8f pv = {};
#pragma unroll
        for (int kt = 0; kt < 2; ++kt) {
          const _Float16* pp = P + l15 * 64 + kt * 32 + 16 * hl;
          v16h pf = cat16(*(const v8h*)pp, *(const v8h*)(pp + 8));
          pv = WMMA_F16(pf, vfrag[dt][kt], pv);
        }
        int d = dt * 16 + l15;
        if (d < 24) {
#pragma unroll
          for (int r = 0; r < 8; ++r)
            ob[(mt * 16 + r + 8 * hl) * 192 + oOff + d] = (_Float16)pv[r];
        }
      }
    }
  }
  __syncthreads();

  // ---- Phase 4: output projections (64x192 @ 192x192), f32 store to global ----
  for (int t = wv; t < 96; t += 8) {
    int g = (t >= 48);
    int tl = t - 48 * g;
    int mt = tl / 12, jt = tl % 12;
    const _Float16* A  = g ? sh : xh;                       // ss -> s_out, ax -> x_out
    const _Float16* Bp = wpack + 147456 + (g ? 36864 : 0);
    const float*    bi = g ? b_projs : b_projx;
    size_t obase = (size_t)g * 25165824ull + ibase;         // s_out after x_out
    v8f acc = {};
#pragma unroll
    for (int kt = 0; kt < 6; ++kt) {
      const _Float16* ap = A + (mt * 16 + l15) * 192 + kt * 32 + 16 * hl;
      v16h af = cat16(*(const v8h*)ap, *(const v8h*)(ap + 8));
      const _Float16* bp = Bp + (((kt * 12 + jt) * 32 + lane) << 4);
      v16h bf = cat16(*(const v8h*)bp, *(const v8h*)(bp + 8));
      acc = WMMA_F16(af, bf, acc);
    }
    float bj = bi[(jt << 4) + l15];
#pragma unroll
    for (int r = 0; r < 8; ++r)
      out[obase + (size_t)((mt * 16 + r + 8 * hl) * 192 + (jt << 4) + l15)] = acc[r] + bj;
  }
}

extern "C" void kernel_launch(void* const* d_in, const int* in_sizes, int n_in,
                              void* d_out, int out_size, void* d_ws, size_t ws_size,
                              hipStream_t stream) {
  (void)in_sizes; (void)n_in; (void)out_size; (void)ws_size;
  const float* input_x    = (const float*)d_in[0];
  const float* state_x    = (const float*)d_in[1];
  const float* bias_xs    = (const float*)d_in[2];
  const float* bt_cross_s = (const float*)d_in[3];
  const float* bt_self_s  = (const float*)d_in[4];
  const float* bt_cross_x = (const float*)d_in[5];
  const float* bt_self_x  = (const float*)d_in[6];
  const float* W_state    = (const float*)d_in[7];
  const float* b_state    = (const float*)d_in[8];
  const float* W_embed    = (const float*)d_in[9];
  const float* b_embed    = (const float*)d_in[10];
  const float* W_projx    = (const float*)d_in[11];
  const float* b_projx    = (const float*)d_in[12];
  const float* W_projs    = (const float*)d_in[13];
  const float* b_projs    = (const float*)d_in[14];
  _Float16* ws = (_Float16*)d_ws;   // needs 221184 f16 = 432 KB of scratch

  // Repack weights into WMMA-B fragment order (deterministic, re-done every call).
  pack_w_kernel<<<288, 256, 0, stream>>>(W_state, ws,           384);
  pack_w_kernel<<<288, 256, 0, stream>>>(W_embed, ws + 73728,   384);
  pack_w_kernel<<<144, 256, 0, stream>>>(W_projx, ws + 147456,  192);
  pack_w_kernel<<<144, 256, 0, stream>>>(W_projs, ws + 184320,  192);

  swin_attn_kernel<<<2048, 256, 0, stream>>>(input_x, state_x, bias_xs,
      bt_cross_s, bt_self_s, bt_cross_x, bt_self_x,
      b_state, b_embed, b_projx, b_projs, ws, (float*)d_out);
}
